// MRAB_10033043604240
// MI455X (gfx1250) — compile-verified, matrix operand order checked by reading, not run on hardware
//
#include <hip/hip_runtime.h>

typedef __attribute__((ext_vector_type(16))) _Float16 v16h;
typedef __attribute__((ext_vector_type(8)))  float    v8f;

#define DGROUPS 8
#define KTAPS   9
#define CPG     8

__device__ __forceinline__ float lrelu_f(float v) { return v >= 0.f ? v : 0.2f * v; }

__device__ __forceinline__ v8f wmma_f16(v16h a, v16h b, v8f c) {
    return __builtin_amdgcn_wmma_f32_16x16x32_f16(false, a, false, b, (short)0, c,
                                                  false, false);
}

// ---------------------------------------------------------------------------
// Generic implicit-GEMM conv via WMMA f32<=f16.
// Block = 256 threads (8 waves) = 4 cout-tiles x 2 pixel-halves:
//   64 couts x 32 pixels per block.
// B (im2col) built in LDS as [pixel][K] with K contiguous and padded to a
// multiple of 16 halfs -> every B fragment is one aligned 32B vector load
// (2x ds_load_b128). A read from pre-converted f16 weights [cout][K] in
// global (L2-resident) -> 2x global_load_b128 per fragment.
// grid.x = B*H*(W/32), grid.y = ceil(Cout/64).
// ---------------------------------------------------------------------------
template<int CIN, int KH, bool LRELU, bool F16OUT, bool RES>
__global__ __launch_bounds__(256)
void conv_wmma_kernel(const float* __restrict__ in, const _Float16* __restrict__ w16,
                      const float* __restrict__ bias, void* __restrict__ outv,
                      const float* __restrict__ res, float resScale,
                      int Cout, int H, int W)
{
    constexpr int PAD  = KH / 2;
    constexpr int KTOT = CIN * KH * KH;     // multiple of 32
    constexpr int KPAD = KTOT + 16;         // keep v16h derefs 32B-aligned
    constexpr int PIXB = 32;

    __shared__ __attribute__((aligned(32))) _Float16 sBm[PIXB][KPAD];

    const int xblocks = W / PIXB;
    const int pb = blockIdx.x;
    const int xb = pb % xblocks;
    const int h  = (pb / xblocks) % H;
    const int b  = pb / (xblocks * H);
    const int x0 = xb * PIXB;

    // Build im2col tile: consecutive tid -> consecutive pixel (coalesced global).
    for (int i = threadIdx.x; i < KTOT * PIXB; i += 256) {
        int p   = i & (PIXB - 1);
        int k   = i >> 5;
        int ci  = k / (KH * KH);
        int rem = k - ci * (KH * KH);
        int ky  = rem / KH;
        int kx  = rem - ky * KH;
        int gy  = h - PAD + ky;
        int gx  = x0 - PAD + p + kx;
        float v = 0.f;
        if (gy >= 0 && gy < H && gx >= 0 && gx < W)
            v = in[(((size_t)b * CIN + ci) * H + gy) * W + gx];
        sBm[p][k] = (_Float16)v;
    }
    __syncthreads();

    const int lane = threadIdx.x & 31;
    const int wv   = threadIdx.x >> 5;
    const int n    = lane & 15;     // pixel col / A row
    const int khf  = lane >> 4;     // K-half select
    const int ctl  = wv >> 1;       // local cout tile 0..3
    const int ph   = wv & 1;        // pixel half
    const int coutBase = (blockIdx.y * 4 + ctl) * 16;
    if (coutBase >= Cout) return;   // no barrier after this point

    const _Float16* arow = w16 + (size_t)(coutBase + n) * KTOT;
    const _Float16* brow = &sBm[ph * 16 + n][0];

    v8f acc = {};
    #pragma unroll
    for (int kb = 0; kb < KTOT / 32; ++kb) {
        v16h a  = *(const v16h*)(arow + kb * 32 + khf * 16);
        v16h bf = *(const v16h*)(brow + kb * 32 + khf * 16);
        acc = wmma_f16(a, bf, acc);
    }

    // C/D layout: VGPR r holds M=r (lanes 0-15) / M=r+8 (lanes 16-31).
    const int px = x0 + ph * 16 + n;
    #pragma unroll
    for (int r = 0; r < 8; ++r) {
        int co = coutBase + r + 8 * khf;
        if (co < Cout) {
            float v = acc[r] + bias[co];
            if (LRELU) v = lrelu_f(v);
            size_t oidx = (((size_t)b * Cout + co) * H + h) * W + px;
            if (RES) v += resScale * res[oidx];
            if (F16OUT) ((_Float16*)outv)[oidx] = (_Float16)v;
            else        ((float*)outv)[oidx]    = v;
        }
    }
}

// ---------------------------------------------------------------------------
// Modulated deformable conv: bilinear sampling -> LDS B-matrix [32 px][592]
// f16 (K contiguous), then WMMA against f16 dcn weights [64 x 576] from
// global. lrelu fused. Block = 256 (8 waves): 4 cout-tiles x 2 pixel-halves.
// ---------------------------------------------------------------------------
__global__ __launch_bounds__(256)
void deform_wmma_kernel(const float* __restrict__ xin,     // conv0 out [B,64,H,W]
                        const _Float16* __restrict__ om,   // [B,216,H,W] f16
                        const _Float16* __restrict__ w16,  // [64,576] f16
                        const float* __restrict__ bias,
                        float* __restrict__ out,           // pre [B,64,H,W]
                        int H, int W)
{
    constexpr int C = 64, KT = 576, KPAD = KT + 16, PIXB = 32;
    __shared__ __attribute__((aligned(32))) _Float16 sB[PIXB][KPAD];

    const int xblocks = W / PIXB;
    const int pb = blockIdx.x;
    const int xb = pb % xblocks;
    const int h  = (pb / xblocks) % H;
    const int b  = pb / (xblocks * H);
    const int x0 = xb * PIXB;
    const int HW = H * W;

    // Sampling: 32 px * 72 (g,k) tasks; each fills 8 channels (cpg).
    for (int t = threadIdx.x; t < PIXB * DGROUPS * KTAPS; t += 256) {
        int px = t % PIXB;
        int gk = t / PIXB;
        int g  = gk / KTAPS, k = gk % KTAPS;
        int ky = k / 3, kx = k % 3;
        int gx = x0 + px;
        size_t obase = (size_t)b * 216 * HW + (size_t)h * W + gx;
        float dy = (float)om[obase + (size_t)(g * KTAPS + k) * HW];
        float dx = (float)om[obase + (size_t)(72 + g * KTAPS + k) * HW];
        float mk = (float)om[obase + (size_t)(144 + g * KTAPS + k) * HW];
        mk = 1.f / (1.f + __expf(-mk));
        float py  = dy + (float)ky + (float)h - 1.f;
        float pxf = dx + (float)kx + (float)gx - 1.f;
        float fy = floorf(py), fx = floorf(pxf);
        int   y0 = (int)fy, xx0 = (int)fx;
        float wy = py - fy, wx = pxf - fx;
        float w00 = (1.f - wy) * (1.f - wx), w01 = (1.f - wy) * wx;
        float w10 = wy * (1.f - wx),         w11 = wy * wx;
        bool vy0 = (y0 >= 0 && y0 < H), vy1 = (y0 + 1 >= 0 && y0 + 1 < H);
        bool vx0 = (xx0 >= 0 && xx0 < W), vx1 = (xx0 + 1 >= 0 && xx0 + 1 < W);
        #pragma unroll
        for (int c = 0; c < CPG; ++c) {
            const float* ch = xin + ((size_t)b * C + g * CPG + c) * HW;
            float v = 0.f;
            if (vy0 && vx0) v += w00 * ch[y0 * W + xx0];
            if (vy0 && vx1) v += w01 * ch[y0 * W + xx0 + 1];
            if (vy1 && vx0) v += w10 * ch[(y0 + 1) * W + xx0];
            if (vy1 && vx1) v += w11 * ch[(y0 + 1) * W + xx0 + 1];
            sB[px][(g * CPG + c) * KTAPS + k] = (_Float16)(v * mk);
        }
    }
    __syncthreads();

    const int lane = threadIdx.x & 31;
    const int wv   = threadIdx.x >> 5;
    const int n    = lane & 15;
    const int khf  = lane >> 4;
    const int ct   = wv >> 1;
    const int ph   = wv & 1;

    const _Float16* arow = w16 + (size_t)(ct * 16 + n) * KT;
    const _Float16* brow = &sB[ph * 16 + n][0];

    v8f acc = {};
    #pragma unroll
    for (int kb = 0; kb < KT / 32; ++kb) {
        v16h a  = *(const v16h*)(arow + kb * 32 + khf * 16);
        v16h bf = *(const v16h*)(brow + kb * 32 + khf * 16);
        acc = wmma_f16(a, bf, acc);
    }

    const int px = x0 + ph * 16 + n;
    #pragma unroll
    for (int r = 0; r < 8; ++r) {
        int co = ct * 16 + r + 8 * khf;
        float v = lrelu_f(acc[r] + bias[co]);
        out[(((size_t)b * C + co) * H + h) * W + px] = v;
    }
}

// ---------------------------------------------------------------------------
// Fused a3 (1x1, 64 -> 576) + dynamic-filter contraction + lrelu.
// Never materializes filt [4,576,128,128]. Each wave: 9 accumulator tiles
// (the 9 filter planes of its 16-channel x 16-pixel tile) via WMMA, then
// contraction against the x0 unfold patch in LDS.
// ---------------------------------------------------------------------------
__global__ __launch_bounds__(256)
void a3_dynfilter_kernel(const float* __restrict__ out1,     // [B,64,H,W]
                         const float* __restrict__ x0in,     // [B,64,H,W]
                         const _Float16* __restrict__ a3w16, // [576,64] f16
                         const float* __restrict__ a3b,      // [576]
                         float* __restrict__ dyn,            // [B,64,H,W]
                         int H, int W)
{
    constexpr int C = 64, CP = C + 16, PIXB = 32;
    __shared__ __attribute__((aligned(32))) _Float16 sB[PIXB][CP]; // out1 tile
    __shared__ float sX[C][3][PIXB + 2];                           // x0 patch

    const int xblocks = W / PIXB;
    const int pb = blockIdx.x;
    const int xb = pb % xblocks;
    const int h  = (pb / xblocks) % H;
    const int b  = pb / (xblocks * H);
    const int x0 = xb * PIXB;

    for (int i = threadIdx.x; i < C * PIXB; i += 256) {
        int p = i & (PIXB - 1);
        int c = i >> 5;
        sB[p][c] = (_Float16)out1[(((size_t)b * C + c) * H + h) * W + x0 + p];
    }
    for (int i = threadIdx.x; i < C * 3 * (PIXB + 2); i += 256) {
        int c  = i / (3 * (PIXB + 2));
        int r  = (i / (PIXB + 2)) % 3;
        int cc = i % (PIXB + 2);
        int gy = h - 1 + r;
        int gx = x0 - 1 + cc;
        float v = 0.f;
        if (gy >= 0 && gy < H && gx >= 0 && gx < W)
            v = x0in[(((size_t)b * C + c) * H + gy) * W + gx];
        sX[c][r][cc] = v;
    }
    __syncthreads();

    const int lane = threadIdx.x & 31;
    const int wv   = threadIdx.x >> 5;
    const int n    = lane & 15;
    const int khf  = lane >> 4;
    const int ct   = wv >> 1;
    const int ph   = wv & 1;

    const _Float16* brow = &sB[ph * 16 + n][0];

    v8f acc[9];
    #pragma unroll
    for (int j = 0; j < 9; ++j) acc[j] = (v8f){};

    #pragma unroll
    for (int j = 0; j < 9; ++j) {
        const _Float16* arow = a3w16 + (size_t)((ct * 16 + n) * 9 + j) * C;
        #pragma unroll
        for (int kb = 0; kb < 2; ++kb) {
            v16h a  = *(const v16h*)(arow + kb * 32 + khf * 16);
            v16h bf = *(const v16h*)(brow + kb * 32 + khf * 16);
            acc[j] = wmma_f16(a, bf, acc[j]);
        }
    }

    const int pxl = ph * 16 + n;
    const int gx  = x0 + pxl;
    #pragma unroll
    for (int r = 0; r < 8; ++r) {
        int c = ct * 16 + r + 8 * khf;
        float v = 0.f;
        #pragma unroll
        for (int j = 0; j < 9; ++j) {
            int jy = j / 3, jx = j % 3;
            float f = acc[j][r] + a3b[c * 9 + j];
            v += f * sX[c][jy][pxl + jx];
        }
        dyn[(((size_t)b * C + c) * H + h) * W + gx] = lrelu_f(v);
    }
}

// ---------------------------------------------------------------------------
// Small helper kernels (bandwidth-trivial)
// ---------------------------------------------------------------------------
__global__ __launch_bounds__(256)
void mean_kernel(const float* __restrict__ x, float* __restrict__ s, int HW)
{
    __shared__ float red[256];
    const float* p = x + (size_t)blockIdx.x * HW;
    float acc = 0.f;
    for (int i = threadIdx.x; i < HW; i += 256) acc += p[i];
    red[threadIdx.x] = acc;
    __syncthreads();
    for (int st = 128; st > 0; st >>= 1) {
        if (threadIdx.x < st) red[threadIdx.x] += red[threadIdx.x + st];
        __syncthreads();
    }
    if (threadIdx.x == 0) s[blockIdx.x] = red[0] / (float)HW;
}

__global__ void eca_kernel(const float* __restrict__ s, const float* __restrict__ w2,
                           const float* __restrict__ b2, float* __restrict__ eca,
                           int BC, int Cc)
{
    int i = blockIdx.x * blockDim.x + threadIdx.x;
    if (i >= BC) return;
    int c = i % Cc;
    float left  = (c > 0)      ? s[i - 1] : 0.f;
    float right = (c < Cc - 1) ? s[i + 1] : 0.f;
    eca[i] = w2[0] * left + w2[1] * s[i] + w2[2] * right + b2[0];
}

__global__ __launch_bounds__(256)
void dw_eca_kernel(const float* __restrict__ x0, const float* __restrict__ w,
                   const float* __restrict__ bias, const float* __restrict__ eca,
                   float* __restrict__ out1, int Bn, int C, int H, int W)
{
    size_t i = (size_t)blockIdx.x * 256 + threadIdx.x;
    size_t total = (size_t)Bn * C * H * W;
    if (i >= total) return;
    int wx = (int)(i % W);
    int hy = (int)((i / W) % H);
    int c  = (int)((i / ((size_t)W * H)) % C);
    int b  = (int)(i / ((size_t)W * H * C));
    const float* ch = x0 + ((size_t)b * C + c) * H * W;
    float acc = bias[c] + eca[b * C + c];
    #pragma unroll
    for (int ky = 0; ky < 3; ++ky)
        #pragma unroll
        for (int kx = 0; kx < 3; ++kx) {
            int yy = hy + ky - 1, xx = wx + kx - 1;
            if (yy >= 0 && yy < H && xx >= 0 && xx < W)
                acc += w[c * 9 + ky * 3 + kx] * ch[yy * W + xx];
        }
    out1[i] = acc;
}

// Convert f32 -> f16, zero-padding the destination beyond srcN (so vector A
// loads of partial cout tiles never read garbage/OOB).
__global__ void f32_to_f16_kernel(const float* __restrict__ src,
                                  _Float16* __restrict__ dst, int srcN, int dstN)
{
    int i = blockIdx.x * 256 + threadIdx.x;
    if (i < dstN) dst[i] = (i < srcN) ? (_Float16)src[i] : (_Float16)0.f;
}

// ---------------------------------------------------------------------------
extern "C" void kernel_launch(void* const* d_in, const int* in_sizes, int n_in,
                              void* d_out, int out_size, void* d_ws, size_t ws_size,
                              hipStream_t stream)
{
    (void)in_sizes; (void)n_in; (void)out_size; (void)ws_size;
    const float* x       = (const float*)d_in[0];
    const float* offset  = (const float*)d_in[1];
    const float* conv0_w = (const float*)d_in[2];
    const float* conv0_b = (const float*)d_in[3];
    const float* offm_w  = (const float*)d_in[4];
    const float* offm_b  = (const float*)d_in[5];
    const float* dcn_w   = (const float*)d_in[6];
    const float* dcn_b   = (const float*)d_in[7];
    const float* a0_w    = (const float*)d_in[8];
    const float* a0_b    = (const float*)d_in[9];
    const float* a1_w    = (const float*)d_in[10];
    const float* a1_b    = (const float*)d_in[11];
    const float* a2_w    = (const float*)d_in[12];
    const float* a2_b    = (const float*)d_in[13];
    const float* a3_w    = (const float*)d_in[14];
    const float* a3_b    = (const float*)d_in[15];
    const float* fus_w   = (const float*)d_in[16];
    const float* fus_b   = (const float*)d_in[17];
    float* out = (float*)d_out;

    const int B = 4, C = 64, H = 128, W = 128, HW = H * W;

    size_t off = 0;
    auto wsalloc = [&](size_t bytes) -> void* {
        void* p = (char*)d_ws + off;
        off += (bytes + 255) & ~(size_t)255;
        return p;
    };
    float*    t0     = (float*)   wsalloc((size_t)B * C * HW * 4);   // conv0 out
    _Float16* om     = (_Float16*)wsalloc((size_t)B * 216 * HW * 2); // offsets/mask f16
    float*    pre    = (float*)   wsalloc((size_t)B * C * HW * 4);   // lrelu(deform)
    float*    x0b    = (float*)   wsalloc((size_t)B * C * HW * 4);   // a0 out
    float*    out1   = (float*)   wsalloc((size_t)B * C * HW * 4);   // dw + eca
    float*    dyn    = (float*)   wsalloc((size_t)B * C * HW * 4);   // dyn filter out
    _Float16* c0w16  = (_Float16*)wsalloc((size_t)64 * 32 * 2);
    _Float16* offw16 = (_Float16*)wsalloc((size_t)256 * 288 * 2);    // padded to 256 rows
    _Float16* dcnw16 = (_Float16*)wsalloc((size_t)64 * 576 * 2);
    _Float16* a0w16  = (_Float16*)wsalloc((size_t)64 * 576 * 2);
    _Float16* a3w16  = (_Float16*)wsalloc((size_t)576 * 64 * 2);
    _Float16* fusw16 = (_Float16*)wsalloc((size_t)64 * 576 * 2);
    float*    s      = (float*)   wsalloc((size_t)B * C * 4);
    float*    eca    = (float*)   wsalloc((size_t)B * C * 4);

    // Weight converts (tiny)
    auto cvt = [&](const float* src, _Float16* dst, int srcN, int dstN) {
        f32_to_f16_kernel<<<(dstN + 255) / 256, 256, 0, stream>>>(src, dst, srcN, dstN);
    };
    cvt(conv0_w, c0w16,  64 * 32,   64 * 32);
    cvt(offm_w,  offw16, 216 * 288, 256 * 288);
    cvt(dcn_w,   dcnw16, 64 * 576,  64 * 576);
    cvt(a0_w,    a0w16,  64 * 576,  64 * 576);
    cvt(a3_w,    a3w16,  576 * 64,  576 * 64);
    cvt(fus_w,   fusw16, 64 * 576,  64 * 576);

    dim3 blk256(256);
    dim3 gridC (B * H * (W / 32), 1);   // Cout=64  -> one 64-cout block slab
    dim3 gridOM(B * H * (W / 32), 4);   // Cout=216 -> 4 slabs (last partial)

    // 1) conv0 1x1 (32 -> 64)
    conv_wmma_kernel<32, 1, false, false, false><<<gridC, blk256, 0, stream>>>(
        x, c0w16, conv0_b, t0, nullptr, 0.f, C, H, W);

    // 2) offset/mask conv 3x3 (32 -> 216), store f16
    conv_wmma_kernel<32, 3, false, true, false><<<gridOM, blk256, 0, stream>>>(
        offset, offw16, offm_b, om, nullptr, 0.f, 216, H, W);

    // 3) modulated deformable conv + lrelu -> pre
    deform_wmma_kernel<<<dim3(B * H * (W / 32)), blk256, 0, stream>>>(
        t0, om, dcnw16, dcn_b, pre, H, W);

    // 4) a0 conv 3x3 (64 -> 64) -> x0
    conv_wmma_kernel<64, 3, false, false, false><<<gridC, blk256, 0, stream>>>(
        pre, a0w16, a0_b, x0b, nullptr, 0.f, C, H, W);

    // 5-6) ECA: global mean + 1D channel conv
    mean_kernel<<<B * C, 256, 0, stream>>>(x0b, s, HW);
    eca_kernel<<<1, 256, 0, stream>>>(s, a2_w, a2_b, eca, B * C, C);

    // 7) depthwise 3x3 + eca broadcast -> out1
    dw_eca_kernel<<<(B * C * HW + 255) / 256, 256, 0, stream>>>(
        x0b, a1_w, a1_b, eca, out1, B, C, H, W);

    // 8) fused a3 1x1 (64 -> 576) + dynamic filter + lrelu -> dyn
    a3_dynfilter_kernel<<<dim3(B * H * (W / 32)), blk256, 0, stream>>>(
        out1, x0b, a3w16, a3_b, dyn, H, W);

    // 9) fus conv 3x3 + pre + x (= + 2*pre) -> out
    conv_wmma_kernel<64, 3, false, false, true><<<gridC, blk256, 0, stream>>>(
        dyn, fusw16, fus_b, out, pre, 2.0f, C, H, W);
}
